// Attention_50156628082997
// MI455X (gfx1250) — matrix operand order repaired and checked by reading, not compiled, not run on hardware
//
#include <hip/hip_runtime.h>
#include <hip/hip_bf16.h>

// ---------------------------------------------------------------------------
// Types
// ---------------------------------------------------------------------------
typedef __bf16 bhalf;
typedef __attribute__((ext_vector_type(16))) __bf16 v16bf;
typedef __attribute__((ext_vector_type(8)))  __bf16 v8bf;
typedef __attribute__((ext_vector_type(8)))  float  v8f;

#define DIM        4096
#define N_HEADS    32
#define N_KV_HEADS 8
#define HEAD_DIM   128
#define BATCH      2
#define SEQ        2048
#define ROWS       (BATCH * SEQ)          // 4096 token rows

__device__ inline unsigned short f32_to_bf16_bits(float f) {
    unsigned int u = __builtin_bit_cast(unsigned int, f);
    unsigned int r = u + 0x7FFFu + ((u >> 16) & 1u);   // round-to-nearest-even
    return (unsigned short)(r >> 16);
}
__device__ inline bhalf f32_to_bf16(float f) {
    unsigned short h = f32_to_bf16_bits(f);
    return __builtin_bit_cast(bhalf, h);
}
__device__ inline v8f v8f_zero() {
    v8f r;
#pragma unroll
    for (int i = 0; i < 8; ++i) r[i] = 0.0f;
    return r;
}

// ---------------------------------------------------------------------------
// WMMA fragment helpers (ISA 7.12.2 layouts, wave32)
// A (16x32 bf16): lane L holds row L&15; lanes<16: K {0..7,16..23}, lanes>=16:
// K {8..15,24..31}  -> two 16B chunks at +khalf and +khalf+16.
// B (32x16 bf16): lane L holds col L&15; same K order as A
// -> two 16B chunks at +khalf and +khalf+16.
// ---------------------------------------------------------------------------
__device__ inline v16bf load_a_frag(const bhalf* base, int row0, int ld, int k0, int lane) {
    const bhalf* p = base + (size_t)(row0 + (lane & 15)) * ld + k0 + ((lane & 16) ? 8 : 0);
    union { v16bf v; v8bf h[2]; } u;
    u.h[0] = *(const v8bf*)(p);
    u.h[1] = *(const v8bf*)(p + 16);
    return u.v;
}
__device__ inline v16bf load_b_frag(const bhalf* base, int row0, int ld, int k0, int lane) {
    const bhalf* p = base + (size_t)(row0 + (lane & 15)) * ld + k0 + ((lane & 16) ? 8 : 0);
    union { v16bf v; v8bf h[2]; } u;
    u.h[0] = *(const v8bf*)(p);
    u.h[1] = *(const v8bf*)(p + 16);
    return u.v;
}
__device__ inline v8f wmma_bf16(v16bf a, v16bf b, v8f c) {
    return __builtin_amdgcn_wmma_f32_16x16x32_bf16(false, a, false, b, (short)0, c,
                                                   false, false);
}

// ---------------------------------------------------------------------------
// fp32 -> bf16 conversion
// ---------------------------------------------------------------------------
__global__ void cvt_f32_bf16(const float* __restrict__ in, bhalf* __restrict__ out, int n) {
    int i = blockIdx.x * blockDim.x + threadIdx.x;
    if (i < n) out[i] = f32_to_bf16(in[i]);
}

// ---------------------------------------------------------------------------
// C[M,N] = A[M,K] * B[N,K]^T   (A,B bf16 row-major, C fp32)
// 256 thr = 8 waves arranged 2(M) x 4(N); block tile 64(M) x 256(N);
// wave tile 32x64 -> 8 WMMA accumulators, 6 fragment loads per K-step
// (~22 FLOP/B from L1, vs 16 with a 32x32 wave tile).
// ---------------------------------------------------------------------------
__global__ __launch_bounds__(256) void gemm_bf16_nt(
    const bhalf* __restrict__ A, const bhalf* __restrict__ Bw,
    float* __restrict__ C, int M, int N, int K)
{
    int lane = threadIdx.x & 31;
    int wid  = threadIdx.x >> 5;
    int wm   = wid & 1;        // 2 waves along M
    int wn   = wid >> 1;       // 4 waves along N
    int m0 = blockIdx.y * 64  + wm * 32;
    int n0 = blockIdx.x * 256 + wn * 64;

    const bhalf* Ab = A  + (size_t)m0 * K;
    const bhalf* Bb = Bw + (size_t)n0 * K;

    v8f acc[2][4];
#pragma unroll
    for (int mi = 0; mi < 2; ++mi)
#pragma unroll
        for (int ni = 0; ni < 4; ++ni) acc[mi][ni] = v8f_zero();

    for (int k0 = 0; k0 < K; k0 += 32) {
        if (k0 + 64 < K) {   // L2 is 192MB: prefetch next K-slab, no LDS staging
            __builtin_prefetch(Ab + (size_t)(lane & 15) * K + k0 + 64, 0, 1);
            __builtin_prefetch(Bb + (size_t)(lane & 15) * K + k0 + 64, 0, 1);
        }
        v16bf a0 = load_a_frag(Ab, 0,  K, k0, lane);
        v16bf a1 = load_a_frag(Ab, 16, K, k0, lane);
#pragma unroll
        for (int ni = 0; ni < 4; ++ni) {
            v16bf b = load_b_frag(Bb, 16 * ni, K, k0, lane);
            acc[0][ni] = wmma_bf16(a0, b, acc[0][ni]);
            acc[1][ni] = wmma_bf16(a1, b, acc[1][ni]);
        }
    }

    int roff = (lane & 16) >> 1;   // +8 rows for upper half-wave
    int col  = lane & 15;
#pragma unroll
    for (int mi = 0; mi < 2; ++mi) {
#pragma unroll
        for (int ni = 0; ni < 4; ++ni) {
#pragma unroll
            for (int j = 0; j < 8; ++j) {
                int m = m0 + 16 * mi + j + roff;
                C[(size_t)m * N + n0 + 16 * ni + col] = acc[mi][ni][j];
            }
        }
    }
}

// ---------------------------------------------------------------------------
// RoPE for Q: qf [ROWS][4096] fp32 -> qb [B][32][T][128] bf16
// ---------------------------------------------------------------------------
__global__ void rope_q_kernel(const float* __restrict__ qf,
                              const float* __restrict__ cosT,
                              const float* __restrict__ sinT,
                              bhalf* __restrict__ qb)
{
    int idx = blockIdx.x * blockDim.x + threadIdx.x;   // B*T*32*64 = 2^23
    int i = idx & 63;
    int h = (idx >> 6) & 31;
    int t = (idx >> 11) & 2047;
    int b = idx >> 22;
    const float* src = qf + (size_t)(b * SEQ + t) * DIM + h * HEAD_DIM + 2 * i;
    float xr = src[0], xi = src[1];
    float c = cosT[t * 64 + i], s = sinT[t * 64 + i];
    bhalf* dst = qb + ((size_t)(b * N_HEADS + h) * SEQ + t) * HEAD_DIM + 2 * i;
    dst[0] = f32_to_bf16(xr * c - xi * s);
    dst[1] = f32_to_bf16(xr * s + xi * c);
}

// ---------------------------------------------------------------------------
// RoPE for K + pack/transpose V:
//   kf/vf [ROWS][1024] fp32 -> kb [B][8][T][128] bf16, vt [B][8][128][T] bf16
// ---------------------------------------------------------------------------
__global__ void rope_kv_kernel(const float* __restrict__ kf,
                               const float* __restrict__ vf,
                               const float* __restrict__ cosT,
                               const float* __restrict__ sinT,
                               bhalf* __restrict__ kb,
                               bhalf* __restrict__ vt)
{
    int idx = blockIdx.x * blockDim.x + threadIdx.x;   // B*T*8*64 = 2^21
    int i   = idx & 63;
    int kvh = (idx >> 6) & 7;
    int t   = (idx >> 9) & 2047;
    int b   = idx >> 20;

    size_t src_off = (size_t)(b * SEQ + t) * (N_KV_HEADS * HEAD_DIM) + kvh * HEAD_DIM + 2 * i;
    float kr = kf[src_off], ki = kf[src_off + 1];
    float c = cosT[t * 64 + i], s = sinT[t * 64 + i];
    bhalf* kd = kb + ((size_t)(b * N_KV_HEADS + kvh) * SEQ + t) * HEAD_DIM + 2 * i;
    kd[0] = f32_to_bf16(kr * c - ki * s);
    kd[1] = f32_to_bf16(kr * s + ki * c);

    float v0 = vf[src_off], v1 = vf[src_off + 1];
    size_t vt_base = ((size_t)(b * N_KV_HEADS + kvh) * HEAD_DIM + 2 * i) * SEQ + t;
    vt[vt_base]       = f32_to_bf16(v0);
    vt[vt_base + SEQ] = f32_to_bf16(v1);
}

// ---------------------------------------------------------------------------
// Flash attention (non-causal; reference never applies mask).
// One wave = one 16-query tile, streams keys 32 at a time.
//   q  [B][32][T][128]  (kv head = h % 8 per jnp.tile semantics)
//   k  [B][8][T][128]
//   vt [B][8][128][T]   (V transposed)
//   ctx [B][T][4096] bf16
// ---------------------------------------------------------------------------
__global__ __launch_bounds__(256) void flash_attn(
    const bhalf* __restrict__ q, const bhalf* __restrict__ k,
    const bhalf* __restrict__ vt, bhalf* __restrict__ ctx)
{
    __shared__ __align__(16) unsigned short lds_p[8][16][32];  // per-wave P tile

    int lane = threadIdx.x & 31;
    int wid  = threadIdx.x >> 5;
    int b = blockIdx.z, h = blockIdx.y;
    int t0 = blockIdx.x * 128 + wid * 16;
    int kvh = h & (N_KV_HEADS - 1);

    const bhalf* qbase = q  + ((size_t)(b * N_HEADS + h)   * SEQ + t0) * HEAD_DIM;
    const bhalf* kbase = k  + ((size_t)(b * N_KV_HEADS + kvh) * SEQ) * HEAD_DIM;
    const bhalf* vbase = vt + ((size_t)(b * N_KV_HEADS + kvh) * HEAD_DIM) * SEQ;

    v16bf qa[4];
#pragma unroll
    for (int kk = 0; kk < 4; ++kk)
        qa[kk] = load_a_frag(qbase, 0, HEAD_DIM, 32 * kk, lane);

    v8f o[8];
#pragma unroll
    for (int di = 0; di < 8; ++di) o[di] = v8f_zero();
    float mrow[8], lrow[8];
#pragma unroll
    for (int j = 0; j < 8; ++j) { mrow[j] = -1e30f; lrow[j] = 0.0f; }

    const float scl = 0.088388347648318447f;   // 1/sqrt(128)
    unsigned short* lp = &lds_p[wid][0][0];
    int prow = (lane & 16) >> 1;               // +8 for upper half-wave
    int pcol = lane & 15;

    for (int s0 = 0; s0 < SEQ; s0 += 32) {
        // ---- S = Q * K^T for 16q x 32k --------------------------------
        v8f c0 = v8f_zero(), c1 = v8f_zero();
#pragma unroll
        for (int kk = 0; kk < 4; ++kk) {
            v16bf b0 = load_b_frag(kbase, s0,      HEAD_DIM, 32 * kk, lane);
            v16bf b1 = load_b_frag(kbase, s0 + 16, HEAD_DIM, 32 * kk, lane);
            c0 = wmma_bf16(qa[kk], b0, c0);
            c1 = wmma_bf16(qa[kk], b1, c1);
        }
        // ---- online softmax ------------------------------------------
#pragma unroll
        for (int j = 0; j < 8; ++j) {
            float s0v = c0[j] * scl;
            float s1v = c1[j] * scl;
            float mx = fmaxf(s0v, s1v);
            mx = fmaxf(mx, __shfl_xor(mx, 1, 32));
            mx = fmaxf(mx, __shfl_xor(mx, 2, 32));
            mx = fmaxf(mx, __shfl_xor(mx, 4, 32));
            mx = fmaxf(mx, __shfl_xor(mx, 8, 32));
            float mnew = fmaxf(mrow[j], mx);
            float corr = __expf(mrow[j] - mnew);
            float p0 = __expf(s0v - mnew);
            float p1 = __expf(s1v - mnew);
            float sum = p0 + p1;
            sum += __shfl_xor(sum, 1, 32);
            sum += __shfl_xor(sum, 2, 32);
            sum += __shfl_xor(sum, 4, 32);
            sum += __shfl_xor(sum, 8, 32);
            lrow[j] = lrow[j] * corr + sum;
            mrow[j] = mnew;
#pragma unroll
            for (int di = 0; di < 8; ++di) o[di][j] *= corr;
            lp[(prow + j) * 32 + pcol]      = f32_to_bf16_bits(p0);
            lp[(prow + j) * 32 + 16 + pcol] = f32_to_bf16_bits(p1);
        }
        asm volatile("s_wait_dscnt 0" ::: "memory");
        v16bf pa = load_a_frag((const bhalf*)lp, 0, 32, 0, lane);
        // ---- O += P * V  (Vt rows are d, K-dim is s) ------------------
#pragma unroll
        for (int di = 0; di < 8; ++di) {
            v16bf bv = load_b_frag(vbase, 16 * di, SEQ, s0, lane);
            o[di] = wmma_bf16(pa, bv, o[di]);
        }
    }

    // ---- normalize + scatter into ctx[b][t][h*128+d] ------------------
    bhalf* cbase = ctx + (size_t)b * SEQ * DIM + (size_t)h * HEAD_DIM;
#pragma unroll
    for (int j = 0; j < 8; ++j) {
        float inv = 1.0f / lrow[j];
        int t = t0 + j + prow;
#pragma unroll
        for (int di = 0; di < 8; ++di) {
            cbase[(size_t)t * DIM + 16 * di + (lane & 15)] = f32_to_bf16(o[di][j] * inv);
        }
    }
}

// ---------------------------------------------------------------------------
// Launch
// ---------------------------------------------------------------------------
extern "C" void kernel_launch(void* const* d_in, const int* in_sizes, int n_in,
                              void* d_out, int out_size, void* d_ws, size_t ws_size,
                              hipStream_t stream) {
    (void)in_sizes; (void)n_in; (void)out_size; (void)ws_size;

    const float* x  = (const float*)d_in[0];
    const float* wq = (const float*)d_in[1];
    const float* wk = (const float*)d_in[2];
    const float* wv = (const float*)d_in[3];
    const float* wo = (const float*)d_in[4];
    const float* fc = (const float*)d_in[5];
    const float* fs = (const float*)d_in[6];
    float* out = (float*)d_out;
    char*  ws  = (char*)d_ws;

    // Workspace layout (aliased regions noted; stream order makes it safe)
    bhalf* xb  = (bhalf*)(ws + 0);            //  33.5MB  (reused as qb)
    bhalf* wqb = (bhalf*)(ws + 33554432ull);  //  33.5MB
    bhalf* wkb = (bhalf*)(ws + 67108864ull);  //   8.4MB  (reused as kb)
    bhalf* wvb = (bhalf*)(ws + 75497472ull);  //   8.4MB  (reused as vt)
    bhalf* wob = (bhalf*)(ws + 83886080ull);  //  33.5MB
    float* qf  = (float*)(ws + 117440512ull); //  67.1MB  (reused as ctx)
    float* kf  = (float*)(ws + 184549376ull); //  16.8MB
    float* vf  = (float*)(ws + 201326592ull); //  16.8MB  -> total 218.1MB
    bhalf* qb   = xb;
    bhalf* kb   = wkb;
    bhalf* vt   = wvb;
    bhalf* ctxb = (bhalf*)qf;

    // 1) fp32 -> bf16 conversions
    cvt_f32_bf16<<<65536, 256, 0, stream>>>(x,  xb,  ROWS * DIM);
    cvt_f32_bf16<<<65536, 256, 0, stream>>>(wq, wqb, DIM * DIM);
    cvt_f32_bf16<<<16384, 256, 0, stream>>>(wk, wkb, N_KV_HEADS * HEAD_DIM * DIM);
    cvt_f32_bf16<<<16384, 256, 0, stream>>>(wv, wvb, N_KV_HEADS * HEAD_DIM * DIM);
    cvt_f32_bf16<<<65536, 256, 0, stream>>>(wo, wob, DIM * DIM);

    // 2) QKV projections (WMMA bf16 -> f32); block tile 64(M) x 256(N)
    gemm_bf16_nt<<<dim3(16, 64), 256, 0, stream>>>(xb, wqb, qf, ROWS, DIM, DIM);
    gemm_bf16_nt<<<dim3(4,  64), 256, 0, stream>>>(xb, wkb, kf, ROWS, N_KV_HEADS * HEAD_DIM, DIM);
    gemm_bf16_nt<<<dim3(4,  64), 256, 0, stream>>>(xb, wvb, vf, ROWS, N_KV_HEADS * HEAD_DIM, DIM);

    // 3) RoPE + pack to attention layouts
    rope_q_kernel<<<32768, 256, 0, stream>>>(qf, fc, fs, qb);
    rope_kv_kernel<<<8192, 256, 0, stream>>>(kf, vf, fc, fs, kb, vt);

    // 4) Flash attention (grid: T/128 q-tiles x heads x batch)
    flash_attn<<<dim3(SEQ / 128, N_HEADS, BATCH), 256, 0, stream>>>(qb, kb, vt, ctxb);

    // 5) Output projection -> d_out (fp32)
    gemm_bf16_nt<<<dim3(16, 64), 256, 0, stream>>>(ctxb, wob, out, ROWS, DIM, DIM);
}
